// WaveRNNONNX_14224931684623
// MI455X (gfx1250) — matrix operand order repaired and checked by reading, not compiled, not run on hardware
//
#include <hip/hip_runtime.h>

// ---------------------------------------------------------------------------
// Types for CDNA5 WMMA (wave32): bf16 A/B fragments, f32 accumulator.
// ---------------------------------------------------------------------------
typedef __attribute__((ext_vector_type(16))) __bf16 bf16x16;
typedef __attribute__((ext_vector_type(8)))  __bf16 bf16x8;
typedef __attribute__((ext_vector_type(8)))  float  f32x8;

union V16 { bf16x16 v; bf16x8 h[2]; };

__device__ __forceinline__ __bf16 f2bf(float f) {
  union { float f; unsigned u; } cv; cv.f = f;
  unsigned u = cv.u;
  u += 0x7FFFu + ((u >> 16) & 1u);               // round-to-nearest-even
  union { unsigned short s; __bf16 b; } cb; cb.s = (unsigned short)(u >> 16);
  return cb.b;
}

// A fragment: 16x32 bf16. Lane l: row m = l&15; holds 16 consecutive K
// at kb = (l>>4)*16.  base must already include row*lda + kb.
__device__ __forceinline__ bf16x16 load_a16x32(const __bf16* base, int k) {
  V16 t;
  t.h[0] = *(const bf16x8*)(base + k);
  t.h[1] = *(const bf16x8*)(base + k + 8);
  return t.v;
}

// B fragment: 32x16 bf16 from row-major W[N][K] (B(k,n) = W[n][k]).
// Lane l: col n = l&15; holds 16 consecutive K at kb = (l>>4)*16.
// base must already include n*ldw + kb.
__device__ __forceinline__ bf16x16 load_b32x16(const __bf16* base, int k) {
  V16 t;
  t.h[0] = *(const bf16x8*)(base + k);
  t.h[1] = *(const bf16x8*)(base + k + 8);
  return t.v;
}

__device__ __forceinline__ f32x8 wmma_bf16(bf16x16 a, bf16x16 b, f32x8 c) {
  return __builtin_amdgcn_wmma_f32_16x16x32_bf16(
      /*neg_a=*/false, a, /*neg_b=*/false, b,
      /*c_mod=*/(short)0, c, /*reuse_a=*/false, /*reuse_b=*/false);
}

// ---------------------------------------------------------------------------
// Prep kernels: f32 -> bf16 casts / concats (weights stay L2-resident after).
// ---------------------------------------------------------------------------
__global__ void cast_pad_bf16_k(const float* __restrict__ src, __bf16* __restrict__ dst,
                                int K, int Kpad, int total) {
  int idx = blockIdx.x * blockDim.x + threadIdx.x;
  if (idx >= total) return;
  int r = idx / Kpad, c = idx - r * Kpad;
  float v = (c < K) ? src[(size_t)r * K + c] : 0.0f;
  dst[idx] = f2bf(v);
}

// X0 = [x | m_t | a1_t | zero-pad] as B x 128 bf16
__global__ void build_x0_k(const float* __restrict__ x, const float* __restrict__ mel,
                           const float* __restrict__ a1, __bf16* __restrict__ dst) {
  int idx = blockIdx.x * blockDim.x + threadIdx.x;   // B*128 threads
  int m = idx >> 7, c = idx & 127;
  float v = 0.0f;
  if (c == 0)        v = x[m];
  else if (c < 81)   v = mel[(size_t)m * 80 + (c - 1)];
  else if (c < 113)  v = a1[(size_t)m * 32 + (c - 81)];
  dst[idx] = f2bf(v);
}

// write bf16(aux[B x 32]) into dst columns [off, off+32) with leading dim ld
__global__ void concat_aux_k(const float* __restrict__ aux, __bf16* __restrict__ dst,
                             int ld, int off) {
  int idx = blockIdx.x * blockDim.x + threadIdx.x;   // B*32 threads
  int m = idx >> 5, c = idx & 31;
  dst[(size_t)m * ld + off + c] = f2bf(aux[idx]);
}

// ---------------------------------------------------------------------------
// Generic GEMM: C[B x N] = A_bf16[B x K] @ W_bf16[N x K]^T + bias, optional
// ReLU; writes f32 and/or bf16 (for chaining). One wave -> 32(M) x 64(N):
// 2 A fragments x 4 B fragments -> 8 WMMAs per K-step (B reused across M,
// A reused across N: ~21 FLOP per L2 byte). No LDS: all operands are
// L2-resident (192 MB).
// ---------------------------------------------------------------------------
__global__ __launch_bounds__(256) void gemm_bf16_wmma(
    const __bf16* __restrict__ A, int lda,
    const __bf16* __restrict__ W, int ldw,
    const float* __restrict__ bias,
    int N, int K, int relu,
    float* __restrict__ outF, int ldoF,
    __bf16* __restrict__ outH, int ldoH)
{
  const int lane = threadIdx.x & 31;
  const int wid  = blockIdx.x * (blockDim.x >> 5) + (threadIdx.x >> 5);
  const int tilesN = N >> 6;
  const int mt = wid / tilesN;
  const int nt = wid - mt * tilesN;
  const int m0 = mt << 5;                    // 32 rows per wave
  const int n0 = nt << 6;                    // 64 cols per wave

  const __bf16* Ap0 = A + (size_t)(m0 + (lane & 15)) * lda + ((lane >> 4) << 4);
  const __bf16* Ap1 = Ap0 + (size_t)16 * lda;
  const __bf16* Wp  = W + (size_t)(n0 + (lane & 15)) * ldw + ((lane >> 4) << 4);
  const size_t ws16 = (size_t)16 * ldw;

  f32x8 acc00 = {}, acc01 = {}, acc02 = {}, acc03 = {};
  f32x8 acc10 = {}, acc11 = {}, acc12 = {}, acc13 = {};
  for (int k = 0; k < K; k += 32) {
    bf16x16 a0 = load_a16x32(Ap0, k);
    bf16x16 a1 = load_a16x32(Ap1, k);
    bf16x16 b0 = load_b32x16(Wp, k);
    acc00 = wmma_bf16(a0, b0, acc00);
    acc10 = wmma_bf16(a1, b0, acc10);
    bf16x16 b1 = load_b32x16(Wp + ws16, k);
    acc01 = wmma_bf16(a0, b1, acc01);
    acc11 = wmma_bf16(a1, b1, acc11);
    bf16x16 b2 = load_b32x16(Wp + 2 * ws16, k);
    acc02 = wmma_bf16(a0, b2, acc02);
    acc12 = wmma_bf16(a1, b2, acc12);
    bf16x16 b3 = load_b32x16(Wp + 3 * ws16, k);
    acc03 = wmma_bf16(a0, b3, acc03);
    acc13 = wmma_bf16(a1, b3, acc13);
  }

  const int rowAdd = (lane >> 4) << 3;
  const int ncol   = lane & 15;
  f32x8 accs[2][4] = {{acc00, acc01, acc02, acc03}, {acc10, acc11, acc12, acc13}};
#pragma unroll
  for (int i = 0; i < 2; ++i) {
#pragma unroll
    for (int j = 0; j < 4; ++j) {
      int n = n0 + j * 16 + ncol;
      float bb = bias ? bias[n] : 0.0f;
#pragma unroll
      for (int r = 0; r < 8; ++r) {
        int m = m0 + i * 16 + rowAdd + r;
        float v = accs[i][j][r] + bb;
        if (relu) v = fmaxf(v, 0.0f);
        if (outF) outF[(size_t)m * ldoF + n] = v;
        if (outH) outH[(size_t)m * ldoH + n] = f2bf(v);
      }
    }
  }
}

// ---------------------------------------------------------------------------
// Fused GRU cell: one wave computes 32(M) x 16(N) of h_new. The six gate
// B fragments (gi_r/z/n from Wih, gh_r/z/n from Whh) are each reused by two
// A fragments (6 WMMAs per K-step from 5 fragment loads in each K loop). Epilogue applies
// the GRU nonlinearity and writes h_new (f32), x += h_new (f32 in place),
// and the bf16 input for the next layer. Avoids materializing the B x 1536
// gate tensors entirely.  H is fixed at 512.
// ---------------------------------------------------------------------------
__global__ __launch_bounds__(256) void gru_fused_wmma(
    const __bf16* __restrict__ Xb, int ldx, int Kx,
    const __bf16* __restrict__ Hb,
    const __bf16* __restrict__ Wih, int ldwih,
    const __bf16* __restrict__ Whh,
    const float* __restrict__ bih, const float* __restrict__ bhh,
    const float* __restrict__ hprev,
    float* __restrict__ hout,
    float* __restrict__ xcur,
    __bf16* __restrict__ xnext, int ldxn)
{
  const int H = 512;
  const int lane = threadIdx.x & 31;
  const int wid  = blockIdx.x * (blockDim.x >> 5) + (threadIdx.x >> 5);
  const int tilesN = H >> 4;                 // 32
  const int mt = wid / tilesN;
  const int nt = wid - mt * tilesN;
  const int m0 = mt << 5;                    // 32 rows per wave
  const int n0 = nt << 4;

  f32x8 ir0 = {}, iz0 = {}, in0 = {}, hr0 = {}, hz0 = {}, hn0 = {};
  f32x8 ir1 = {}, iz1 = {}, in1 = {}, hr1 = {}, hz1 = {}, hn1 = {};

  {   // gi = X @ Wih^T  (three gate column blocks: +0, +512, +1024)
    const __bf16* Ap0 = Xb  + (size_t)(m0 + (lane & 15)) * ldx   + ((lane >> 4) << 4);
    const __bf16* Ap1 = Ap0 + (size_t)16 * ldx;
    const __bf16* Wr = Wih + (size_t)(n0 + (lane & 15)) * ldwih + ((lane >> 4) << 4);
    const __bf16* Wz = Wr + (size_t)512  * ldwih;
    const __bf16* Wn = Wr + (size_t)1024 * ldwih;
    for (int k = 0; k < Kx; k += 32) {
      bf16x16 a0 = load_a16x32(Ap0, k);
      bf16x16 a1 = load_a16x32(Ap1, k);
      bf16x16 br = load_b32x16(Wr, k);
      ir0 = wmma_bf16(a0, br, ir0);
      ir1 = wmma_bf16(a1, br, ir1);
      bf16x16 bz = load_b32x16(Wz, k);
      iz0 = wmma_bf16(a0, bz, iz0);
      iz1 = wmma_bf16(a1, bz, iz1);
      bf16x16 bn = load_b32x16(Wn, k);
      in0 = wmma_bf16(a0, bn, in0);
      in1 = wmma_bf16(a1, bn, in1);
    }
  }
  {   // gh = H @ Whh^T
    const __bf16* Ap0 = Hb  + (size_t)(m0 + (lane & 15)) * H + ((lane >> 4) << 4);
    const __bf16* Ap1 = Ap0 + (size_t)16 * H;
    const __bf16* Wr = Whh + (size_t)(n0 + (lane & 15)) * H + ((lane >> 4) << 4);
    const __bf16* Wz = Wr + (size_t)512  * H;
    const __bf16* Wn = Wr + (size_t)1024 * H;
    for (int k = 0; k < H; k += 32) {
      bf16x16 a0 = load_a16x32(Ap0, k);
      bf16x16 a1 = load_a16x32(Ap1, k);
      bf16x16 br = load_b32x16(Wr, k);
      hr0 = wmma_bf16(a0, br, hr0);
      hr1 = wmma_bf16(a1, br, hr1);
      bf16x16 bz = load_b32x16(Wz, k);
      hz0 = wmma_bf16(a0, bz, hz0);
      hz1 = wmma_bf16(a1, bz, hz1);
      bf16x16 bn = load_b32x16(Wn, k);
      hn0 = wmma_bf16(a0, bn, hn0);
      hn1 = wmma_bf16(a1, bn, hn1);
    }
  }

  const int rowAdd = (lane >> 4) << 3;
  const int n = n0 + (lane & 15);
  const float b_ir = bih[n], b_iz = bih[n + 512], b_in = bih[n + 1024];
  const float b_hr = bhh[n], b_hz = bhh[n + 512], b_hn = bhh[n + 1024];

  f32x8 girs[2] = {ir0, ir1}, gizs[2] = {iz0, iz1}, gins[2] = {in0, in1};
  f32x8 ghrs[2] = {hr0, hr1}, ghzs[2] = {hz0, hz1}, ghns[2] = {hn0, hn1};
#pragma unroll
  for (int i = 0; i < 2; ++i) {
#pragma unroll
    for (int r = 0; r < 8; ++r) {
      int m = m0 + i * 16 + rowAdd + r;
      float ir  = girs[i][r] + b_ir;
      float iz  = gizs[i][r] + b_iz;
      float inn = gins[i][r] + b_in;
      float hr  = ghrs[i][r] + b_hr;
      float hz  = ghzs[i][r] + b_hz;
      float hn  = ghns[i][r] + b_hn;
      float rg = 1.0f / (1.0f + __expf(-(ir + hr)));
      float zg = 1.0f / (1.0f + __expf(-(iz + hz)));
      float ng = tanhf(inn + rg * hn);
      size_t idx = (size_t)m * 512 + n;
      float hp   = hprev[idx];
      float hnew = (1.0f - zg) * ng + zg * hp;
      hout[idx] = hnew;
      float xn = xcur[idx] + hnew;
      xcur[idx] = xn;
      xnext[(size_t)m * ldxn + n] = f2bf(xn);
    }
  }
}

// ---------------------------------------------------------------------------
// Launcher
// ---------------------------------------------------------------------------
extern "C" void kernel_launch(void* const* d_in, const int* in_sizes, int n_in,
                              void* d_out, int out_size, void* d_ws, size_t ws_size,
                              hipStream_t stream) {
  (void)in_sizes; (void)n_in; (void)out_size; (void)ws_size;
  constexpr int B = 8192, RNN = 512, NC = 512;

  const float* m_t   = (const float*)d_in[0];
  const float* a1    = (const float*)d_in[1];
  const float* a2    = (const float*)d_in[2];
  const float* a3    = (const float*)d_in[3];
  const float* a4    = (const float*)d_in[4];
  const float* h1    = (const float*)d_in[5];
  const float* h2    = (const float*)d_in[6];
  const float* x     = (const float*)d_in[7];
  const float* I_w   = (const float*)d_in[8];
  const float* I_b   = (const float*)d_in[9];
  const float* r1wih = (const float*)d_in[10];
  const float* r1whh = (const float*)d_in[11];
  const float* r1bih = (const float*)d_in[12];
  const float* r1bhh = (const float*)d_in[13];
  const float* r2wih = (const float*)d_in[14];
  const float* r2whh = (const float*)d_in[15];
  const float* r2bih = (const float*)d_in[16];
  const float* r2bhh = (const float*)d_in[17];
  const float* f1w   = (const float*)d_in[18];
  const float* f1b   = (const float*)d_in[19];
  const float* f2w   = (const float*)d_in[20];
  const float* f2b   = (const float*)d_in[21];
  const float* f3w   = (const float*)d_in[22];
  const float* f3b   = (const float*)d_in[23];

  float* logits = (float*)d_out;                    // (logits, h1n, h2n)
  float* h1n    = logits + (size_t)B * NC;
  float* h2n    = h1n    + (size_t)B * RNN;

  // workspace carving (256B aligned regions), ~62 MB total
  char* base = (char*)d_ws;
  size_t off = 0;
  auto carve = [&](size_t elems, size_t esz) -> void* {
    void* p = base + off;
    off += (elems * esz + 255) & ~(size_t)255;
    return p;
  };
  __bf16* wI   = (__bf16*)carve(512 * 128, 2);
  __bf16* w1ih = (__bf16*)carve(1536 * 512, 2);
  __bf16* w1hh = (__bf16*)carve(1536 * 512, 2);
  __bf16* w2ih = (__bf16*)carve(1536 * 544, 2);
  __bf16* w2hh = (__bf16*)carve(1536 * 512, 2);
  __bf16* wf1  = (__bf16*)carve(512 * 544, 2);
  __bf16* wf2  = (__bf16*)carve(512 * 544, 2);
  __bf16* wf3  = (__bf16*)carve(512 * 512, 2);
  __bf16* h1b  = (__bf16*)carve((size_t)B * 512, 2);
  __bf16* h2b  = (__bf16*)carve((size_t)B * 512, 2);
  __bf16* X0   = (__bf16*)carve((size_t)B * 128, 2);
  float*  xcur = (float*)carve((size_t)B * 512, 4);
  __bf16* bufA = (__bf16*)carve((size_t)B * 544, 2);
  __bf16* bufB = (__bf16*)carve((size_t)B * 544, 2);

  auto cg = [](size_t total) { return (unsigned)((total + 255) / 256); };

  // --- prep: cast weights / states to bf16 (K padded to mult. of 32) -------
  cast_pad_bf16_k<<<cg(512 * 128),  256, 0, stream>>>(I_w,   wI,   113, 128, 512 * 128);
  cast_pad_bf16_k<<<cg(1536 * 512), 256, 0, stream>>>(r1wih, w1ih, 512, 512, 1536 * 512);
  cast_pad_bf16_k<<<cg(1536 * 512), 256, 0, stream>>>(r1whh, w1hh, 512, 512, 1536 * 512);
  cast_pad_bf16_k<<<cg(1536 * 544), 256, 0, stream>>>(r2wih, w2ih, 544, 544, 1536 * 544);
  cast_pad_bf16_k<<<cg(1536 * 512), 256, 0, stream>>>(r2whh, w2hh, 512, 512, 1536 * 512);
  cast_pad_bf16_k<<<cg(512 * 544),  256, 0, stream>>>(f1w,   wf1,  544, 544, 512 * 544);
  cast_pad_bf16_k<<<cg(512 * 544),  256, 0, stream>>>(f2w,   wf2,  544, 544, 512 * 544);
  cast_pad_bf16_k<<<cg(512 * 512),  256, 0, stream>>>(f3w,   wf3,  512, 512, 512 * 512);
  cast_pad_bf16_k<<<cg((size_t)B * 512), 256, 0, stream>>>(h1, h1b, 512, 512, B * 512);
  cast_pad_bf16_k<<<cg((size_t)B * 512), 256, 0, stream>>>(h2, h2b, 512, 512, B * 512);
  build_x0_k<<<cg((size_t)B * 128), 256, 0, stream>>>(x, m_t, a1, X0);

  const unsigned gemmBlocks = (B / 32) * (512 / 64) / 8;   // 256 blocks (8 waves each)
  const unsigned gruBlocks  = (B / 32) * 32 / 8;           // 1024 blocks

  // I: xfeat = X0 @ I_w^T + I_b  -> xcur (f32) and bufA (bf16, ld 512)
  gemm_bf16_wmma<<<gemmBlocks, 256, 0, stream>>>(X0, 128, wI, 128, I_b, 512, 128, 0,
                                                 xcur, 512, bufA, 512);
  // GRU1: h1n -> d_out, xcur += h1n, bf16 x -> bufB cols[0:512) (ld 544)
  gru_fused_wmma<<<gruBlocks, 256, 0, stream>>>(bufA, 512, 512, h1b, w1ih, 512, w1hh,
                                                r1bih, r1bhh, h1, h1n, xcur, bufB, 544);
  concat_aux_k<<<cg((size_t)B * 32), 256, 0, stream>>>(a2, bufB, 544, 512);
  // GRU2
  gru_fused_wmma<<<gruBlocks, 256, 0, stream>>>(bufB, 544, 544, h2b, w2ih, 544, w2hh,
                                                r2bih, r2bhh, h2, h2n, xcur, bufA, 544);
  concat_aux_k<<<cg((size_t)B * 32), 256, 0, stream>>>(a3, bufA, 544, 512);
  // FC1 (ReLU) -> bufB cols[0:512) (ld 544), then concat a4
  gemm_bf16_wmma<<<gemmBlocks, 256, 0, stream>>>(bufA, 544, wf1, 544, f1b, 512, 544, 1,
                                                 nullptr, 0, bufB, 544);
  concat_aux_k<<<cg((size_t)B * 32), 256, 0, stream>>>(a4, bufB, 544, 512);
  // FC2 (ReLU) -> bufA (ld 512)
  gemm_bf16_wmma<<<gemmBlocks, 256, 0, stream>>>(bufB, 544, wf2, 544, f2b, 512, 544, 1,
                                                 nullptr, 0, bufA, 512);
  // FC3 -> logits (f32, d_out)
  gemm_bf16_wmma<<<gemmBlocks, 256, 0, stream>>>(bufA, 512, wf3, 512, f3b, 512, 512, 0,
                                                 logits, 512, nullptr, 0);
}
